// MixtralAttention_49667001811796
// MI455X (gfx1250) — compile-verified, matrix operand order checked by reading, not run on hardware
//
#include <hip/hip_runtime.h>
#include <hip/hip_bf16.h>

// ---------------------------------------------------------------------------
// Problem constants (from reference)
// ---------------------------------------------------------------------------
#define BB 2
#define SS 2048
#define HH 4096
#define NH 32
#define NKV 8
#define DD 128
#define GROUPS (NH / NKV)
#define MAXLEN SS
#define SCALE 0.08838834764831845f  // 128^-0.5

typedef __attribute__((ext_vector_type(16))) __bf16 v16bf;
typedef __attribute__((ext_vector_type(8)))  float  v8f;

#if defined(__AMDGCN__) && __has_builtin(__builtin_amdgcn_tensor_load_to_lds)
#define HAVE_TDM 1
typedef unsigned int u32x4 __attribute__((ext_vector_type(4)));
typedef int          i32x4t __attribute__((ext_vector_type(4)));
typedef int          i32x8t __attribute__((ext_vector_type(8)));
#else
#define HAVE_TDM 0
#endif

__device__ __forceinline__ v8f wmma_bf16(v16bf a, v16bf b, v8f c) {
  return __builtin_amdgcn_wmma_f32_16x16x32_bf16(false, a, false, b,
                                                 (short)0, c, false, false);
}

__device__ __forceinline__ float rowmax16(float x) {
  #pragma unroll
  for (int o = 1; o < 16; o <<= 1) x = fmaxf(x, __shfl_xor(x, o, 32));
  return x;
}
__device__ __forceinline__ float rowsum16(float x) {
  #pragma unroll
  for (int o = 1; o < 16; o <<= 1) x += __shfl_xor(x, o, 32);
  return x;
}

// ---------------------------------------------------------------------------
// Generic bf16-WMMA GEMM:  C[M,N] = A[M,K] * B[K,N]
//   A: fp32 or bf16 (converted to bf16 in LDS), B: fp32 weights,
//   C: bf16 (intermediates) or fp32 (final output).
// Tile: 128x128 per workgroup, 256 threads = 8 waves (2x4), k-step 64.
// ---------------------------------------------------------------------------
#define KSTEP 64

template <typename AT, typename CT>
__global__ __launch_bounds__(256) void gemm_tile128(
    const AT* __restrict__ A, const float* __restrict__ B, CT* __restrict__ C,
    int M, int N, int K) {
  __shared__ __bf16 As[128 * KSTEP];   // [m][k]
  __shared__ __bf16 Bs[128 * KSTEP];   // [n][k]  (k contiguous)

  const int tile_n = blockIdx.x * 128;
  const int tile_m = blockIdx.y * 128;
  const int t = threadIdx.x;
  const int lane = t & 31, wave = t >> 5;
  const int wy = wave >> 2, wx = wave & 3;   // wave grid 2 x 4
  const int lo = lane & 15, hi = lane >> 4;

  v8f acc[4][2];
  #pragma unroll
  for (int i = 0; i < 4; ++i)
    #pragma unroll
    for (int j = 0; j < 2; ++j) acc[i][j] = (v8f){0,0,0,0,0,0,0,0};

  for (int k0 = 0; k0 < K; k0 += KSTEP) {
    __syncthreads();
    // ---- stage A tile: thread copies 32 elements of one row half ----
    {
      const int row = t >> 1, kk = (t & 1) * 32;
      const AT* src = A + (size_t)(tile_m + row) * K + k0 + kk;
      __bf16* dst = &As[row * KSTEP + kk];
      if constexpr (sizeof(AT) == 4) {
        #pragma unroll
        for (int r = 0; r < 8; ++r) {
          const float4 f = *(const float4*)((const float*)src + 4 * r);
          __bf16 h4[4] __attribute__((aligned(8))) = {
              (__bf16)f.x, (__bf16)f.y, (__bf16)f.z, (__bf16)f.w};
          *(uint2*)&dst[4 * r] = *(const uint2*)h4;
        }
      } else {
        #pragma unroll
        for (int r = 0; r < 4; ++r)
          *(uint4*)&dst[8 * r] = *(const uint4*)((const __bf16*)src + 8 * r);
      }
    }
    // ---- stage B tile transposed: thread reads float4 along n ----
    {
      const int kg = t >> 5, n4 = (t & 31) * 4;
      #pragma unroll
      for (int r = 0; r < 8; ++r) {
        const int k = kg * 8 + r;
        const float4 vsrc =
            *(const float4*)(B + (size_t)(k0 + k) * N + tile_n + n4);
        Bs[(n4 + 0) * KSTEP + k] = (__bf16)vsrc.x;
        Bs[(n4 + 1) * KSTEP + k] = (__bf16)vsrc.y;
        Bs[(n4 + 2) * KSTEP + k] = (__bf16)vsrc.z;
        Bs[(n4 + 3) * KSTEP + k] = (__bf16)vsrc.w;
      }
    }
    if (k0 + KSTEP < K) {  // prefetch next A tile row
      __builtin_prefetch(A + (size_t)(tile_m + (t >> 1)) * K + k0 + KSTEP, 0, 1);
    }
    __syncthreads();

    v16bf afrag[4][2], bfrag[2][2];
    #pragma unroll
    for (int i = 0; i < 4; ++i)
      #pragma unroll
      for (int h = 0; h < 2; ++h)
        afrag[i][h] =
            *(const v16bf*)&As[(wy * 64 + i * 16 + lo) * KSTEP + h * 32 + hi * 16];
    #pragma unroll
    for (int j = 0; j < 2; ++j)
      #pragma unroll
      for (int h = 0; h < 2; ++h)
        bfrag[j][h] =
            *(const v16bf*)&Bs[(wx * 32 + j * 16 + lo) * KSTEP + h * 32 + hi * 16];
    #pragma unroll
    for (int i = 0; i < 4; ++i)
      #pragma unroll
      for (int j = 0; j < 2; ++j)
        #pragma unroll
        for (int h = 0; h < 2; ++h)
          acc[i][j] = wmma_bf16(afrag[i][h], bfrag[j][h], acc[i][j]);
  }

  // ---- epilogue: lane holds col=lo, rows v + 8*hi (16 adjacent lanes
  //      write 16 adjacent columns -> coalesced 32B segments) ----
  #pragma unroll
  for (int i = 0; i < 4; ++i)
    #pragma unroll
    for (int j = 0; j < 2; ++j) {
      const int col = tile_n + wx * 32 + j * 16 + lo;
      #pragma unroll
      for (int v = 0; v < 8; ++v) {
        const int row = tile_m + wy * 64 + i * 16 + v + 8 * hi;
        C[(size_t)row * N + col] = (CT)acc[i][j][v];
      }
    }
}

// ---------------------------------------------------------------------------
// RoPE (rotate-half) applied in-place to bf16 Q and K.
// ---------------------------------------------------------------------------
__global__ __launch_bounds__(256) void rope_kernel(
    __bf16* __restrict__ q, __bf16* __restrict__ k,
    const float* __restrict__ cosp, const float* __restrict__ sinp) {
  const size_t idx = (size_t)blockIdx.x * blockDim.x + threadIdx.x;
  const size_t total = (size_t)BB * SS * (NH + NKV) * (DD / 2);
  if (idx >= total) return;
  const int d = idx & 63;
  const int h = (idx >> 6) % (NH + NKV);
  const size_t tok = idx / ((size_t)(NH + NKV) * 64);

  const float c0 = cosp[tok * DD + d],      s0 = sinp[tok * DD + d];
  const float c1 = cosp[tok * DD + d + 64], s1 = sinp[tok * DD + d + 64];

  __bf16* base = (h < NH) ? q + ((size_t)tok * NH + h) * DD
                          : k + ((size_t)tok * NKV + (h - NH)) * DD;
  const float x0 = (float)base[d];
  const float x1 = (float)base[d + 64];
  base[d]      = (__bf16)(x0 * c0 - x1 * s0);
  base[d + 64] = (__bf16)(x1 * c1 + x0 * s1);
}

// ---------------------------------------------------------------------------
// Flash-style causal GQA attention.
// grid = (S/32, NKV, B); block = 256 = 8 waves.
// K-chunk staged into LDS by the Tensor Data Mover when available
// (strided 2D tile: 32 rows x 256B, row stride NKV*D*2 = 2048B).
// ---------------------------------------------------------------------------
__global__ __launch_bounds__(256) void attn_kernel(
    const __bf16* __restrict__ qb, const __bf16* __restrict__ kb,
    const __bf16* __restrict__ vb, __bf16* __restrict__ ob,
    const unsigned char* __restrict__ amask, const int* __restrict__ cidx_p) {
  __shared__ __bf16 Ks[32 * 128];       // [k][d]
  __shared__ __bf16 Vs[128 * 32];       // [d][k] (transposed)
  __shared__ __bf16 Ps[8][16 * 32];     // per-wave P staging (C->A layout)

  const int t = threadIdx.x, lane = t & 31, wave = t >> 5;
  const int lo = lane & 15, hi = lane >> 4;
  const int qbk = blockIdx.x;           // query block (32 rows)
  const int hk  = blockIdx.y;           // kv head
  const int b   = blockIdx.z;
  const int g = wave >> 1, qsub = wave & 1;
  const int hq = hk * GROUPS + g;
  const int qrow_base = qbk * 32 + qsub * 16;
  const int cidx = *cidx_p;

  // Q fragments: 16 rows x 128 d, 4 chunks of K=32
  v16bf qfrag[4];
  {
    const __bf16* qp = qb + (((size_t)b * SS + qrow_base + lo) * NH + hq) * DD;
    #pragma unroll
    for (int c = 0; c < 4; ++c)
      qfrag[c] = *(const v16bf*)(qp + c * 32 + hi * 16);
  }

  v8f o[8];
  #pragma unroll
  for (int dt = 0; dt < 8; ++dt) o[dt] = (v8f){0,0,0,0,0,0,0,0};
  float m[8], lsum[8];
  #pragma unroll
  for (int v = 0; v < 8; ++v) { m[v] = -1e30f; lsum[v] = 0.0f; }

  const int nchunks = qbk + 1;  // causal
  for (int kc = 0; kc < nchunks; ++kc) {
    const int k0 = kc * 32;
    __syncthreads();

    // ---- stage K chunk [k][d] ----
#if HAVE_TDM
    if (wave == 0) {
      const unsigned lds_off = (unsigned)(unsigned long long)&Ks[0];
      const unsigned long long ga = (unsigned long long)(
          kb + (((size_t)b * SS + k0) * NKV + hk) * DD);
      u32x4 g0;
      g0[0] = 1u;                                   // count=1, user descriptor
      g0[1] = lds_off;                              // lds_addr (bytes)
      g0[2] = (unsigned)(ga & 0xFFFFFFFFu);         // global_addr[31:0]
      g0[3] = (unsigned)((ga >> 32) & 0x1FFFFFFu)   // global_addr[56:32]
              | 0x80000000u;                        // type=2 ("image")
      i32x8t g1;
      g1[0] = 0x00010000;           // data_size=1 -> 2 bytes/element
      g1[1] = (int)(128u << 16);    // tensor_dim0 = 128 elements (row)
      g1[2] = (int)(16384u << 16);  // tensor_dim1 = S*NKV rows
      g1[3] = (int)(128u << 16);    // tile_dim0 = 128 elements
      g1[4] = 32;                   // tile_dim1 = 32 rows, tile_dim2 = 0
      g1[5] = 1024;                 // tensor_dim0_stride = NKV*D elements
      g1[6] = 0;
      g1[7] = 0;
      const i32x4t gz = {0, 0, 0, 0};
#if defined(__clang_major__) && __clang_major__ >= 23
      const i32x8t gz8 = {0, 0, 0, 0, 0, 0, 0, 0};
      __builtin_amdgcn_tensor_load_to_lds(g0, g1, gz, gz, gz8, 0);
#else
      __builtin_amdgcn_tensor_load_to_lds(g0, g1, gz, gz, 0);
#endif
    }
#else
    {
      const int kk = t >> 3, dseg = (t & 7) * 16;
      const __bf16* ksrc =
          kb + (((size_t)b * SS + k0 + kk) * NKV + hk) * DD + dseg;
      *(uint4*)&Ks[kk * 128 + dseg]     = *(const uint4*)ksrc;
      *(uint4*)&Ks[kk * 128 + dseg + 8] = *(const uint4*)(ksrc + 8);
    }
#endif
    // ---- stage V chunk transposed [d][k] (TDM cannot transpose) ----
    {
      const int kk = t >> 3, dseg = (t & 7) * 16;
      const __bf16* vsrc =
          vb + (((size_t)b * SS + k0 + kk) * NKV + hk) * DD + dseg;
      #pragma unroll
      for (int j = 0; j < 16; ++j) Vs[(dseg + j) * 32 + kk] = vsrc[j];
    }
#if HAVE_TDM
    if (wave == 0) __builtin_amdgcn_s_wait_tensorcnt(0);
#endif
    __syncthreads();

    // ---- scores: two 16x16 key tiles over D=128 ----
    v8f sacc[2];
    sacc[0] = (v8f){0,0,0,0,0,0,0,0};
    sacc[1] = (v8f){0,0,0,0,0,0,0,0};
    #pragma unroll
    for (int c = 0; c < 4; ++c) {
      v16bf kf0 = *(const v16bf*)&Ks[(lo)      * 128 + c * 32 + hi * 16];
      v16bf kf1 = *(const v16bf*)&Ks[(16 + lo) * 128 + c * 32 + hi * 16];
      sacc[0] = wmma_bf16(qfrag[c], kf0, sacc[0]);
      sacc[1] = wmma_bf16(qfrag[c], kf1, sacc[1]);
    }

    // ---- online softmax ----
    const int key0 = k0 + lo, key1 = k0 + 16 + lo;
    const bool a0 = amask[(size_t)b * MAXLEN + key0] != 0;
    const bool a1 = amask[(size_t)b * MAXLEN + key1] != 0;
    float alpha[8];
    #pragma unroll
    for (int v = 0; v < 8; ++v) {
      const int row = qrow_base + v + 8 * hi;
      float s0 = sacc[0][v] * SCALE;
      float s1 = sacc[1][v] * SCALE;
      if (!a0 || key0 > cidx + row) s0 = -1e30f;
      if (!a1 || key1 > cidx + row) s1 = -1e30f;
      const float mc = rowmax16(fmaxf(s0, s1));
      const float mn = fmaxf(m[v], mc);
      alpha[v] = __expf(m[v] - mn);
      const float e0 = __expf(s0 - mn);
      const float e1 = __expf(s1 - mn);
      lsum[v] = lsum[v] * alpha[v] + rowsum16(e0 + e1);
      m[v] = mn;
      Ps[wave][(v + 8 * hi) * 32 + lo]      = (__bf16)e0;
      Ps[wave][(v + 8 * hi) * 32 + 16 + lo] = (__bf16)e1;
    }
    #pragma unroll
    for (int dt = 0; dt < 8; ++dt)
      #pragma unroll
      for (int v = 0; v < 8; ++v) o[dt][v] *= alpha[v];

    // ---- P (16x32) x V (32x128): 8 WMMAs ----
    v16bf pa = *(const v16bf*)&Ps[wave][lo * 32 + hi * 16];
    #pragma unroll
    for (int dt = 0; dt < 8; ++dt) {
      v16bf vf = *(const v16bf*)&Vs[(dt * 16 + lo) * 32 + hi * 16];
      o[dt] = wmma_bf16(pa, vf, o[dt]);
    }
  }

  // ---- normalize and write bf16 attn output (B,S,NH,D) ----
  float inv[8];
  #pragma unroll
  for (int v = 0; v < 8; ++v) inv[v] = 1.0f / lsum[v];
  #pragma unroll
  for (int dt = 0; dt < 8; ++dt) {
    const int d = dt * 16 + lo;
    #pragma unroll
    for (int v = 0; v < 8; ++v) {
      const int row = qrow_base + v + 8 * hi;
      ob[(((size_t)b * SS + row) * NH + hq) * DD + d] =
          (__bf16)(o[dt][v] * inv[v]);
    }
  }
}

// ---------------------------------------------------------------------------
// Launch: QKV GEMMs -> RoPE -> attention -> output projection.
// Workspace (bf16 intermediates): Q 32MB | K 8MB | V 8MB | attn_out 32MB.
// ---------------------------------------------------------------------------
extern "C" void kernel_launch(void* const* d_in, const int* in_sizes, int n_in,
                              void* d_out, int out_size, void* d_ws,
                              size_t ws_size, hipStream_t stream) {
  const float* hidden = (const float*)d_in[0];
  const unsigned char* amask = (const unsigned char*)d_in[1];
  const float* cosp = (const float*)d_in[2];
  const float* sinp = (const float*)d_in[3];
  const int* cidx = (const int*)d_in[6];
  const float* wq = (const float*)d_in[7];
  const float* wk = (const float*)d_in[8];
  const float* wv = (const float*)d_in[9];
  const float* wo = (const float*)d_in[10];
  float* out = (float*)d_out;

  char* ws = (char*)d_ws;
  __bf16* qws = (__bf16*)(ws);                               // 32 MB
  __bf16* kws = (__bf16*)(ws + (size_t)32 * 1024 * 1024);    //  8 MB
  __bf16* vws = (__bf16*)(ws + (size_t)40 * 1024 * 1024);    //  8 MB
  __bf16* aws = (__bf16*)(ws + (size_t)48 * 1024 * 1024);    // 32 MB

  const int M = BB * SS;  // 4096 tokens

  gemm_tile128<float, __bf16><<<dim3(NH * DD / 128, M / 128), 256, 0, stream>>>(
      hidden, wq, qws, M, NH * DD, HH);
  gemm_tile128<float, __bf16><<<dim3(NKV * DD / 128, M / 128), 256, 0, stream>>>(
      hidden, wk, kws, M, NKV * DD, HH);
  gemm_tile128<float, __bf16><<<dim3(NKV * DD / 128, M / 128), 256, 0, stream>>>(
      hidden, wv, vws, M, NKV * DD, HH);

  {
    const size_t total = (size_t)BB * SS * (NH + NKV) * (DD / 2);
    rope_kernel<<<(unsigned)((total + 255) / 256), 256, 0, stream>>>(
        qws, kws, cosp, sinp);
  }

  attn_kernel<<<dim3(SS / 32, NKV, BB), 256, 0, stream>>>(qws, kws, vws, aws,
                                                          amask, cidx);

  gemm_tile128<__bf16, float><<<dim3(HH / 128, M / 128), 256, 0, stream>>>(
      aws, wo, out, M, HH, NH * DD);
}
